// Model_37331855736897
// MI455X (gfx1250) — matrix operand order repaired and checked by reading, not compiled, not run on hardware
//
#include <hip/hip_runtime.h>

// ---------------------------------------------------------------------------
// CDNA5 (gfx1250) transformer training step.
// All GEMMs: bf16 WMMA (V_WMMA_F32_16X16X32_BF16), f32 accumulate,
// software-pipelined: global_load_b128 (next tile, regs) overlaps WMMA
// (current tile, LDS). Block tile 128x64, 8 waves x (32x32, 4 WMMA/K-step).
// Model constants: DIM=512 HID=2048 VOCAB=32000 HEADS=8 HD=64 B=4 L=64
// Tokens per chunk N = B*L = 256. 4 chunks, 4 blocks.
// ---------------------------------------------------------------------------

typedef __attribute__((ext_vector_type(16))) __bf16 v16bf;
typedef __attribute__((ext_vector_type(8)))  __bf16 v8bf;
typedef __attribute__((ext_vector_type(4)))  __bf16 v4bf;
typedef __attribute__((ext_vector_type(8)))  float  v8f;
typedef __attribute__((ext_vector_type(4)))  float  f32x4;

#define NTOK 256
#define DIMC 512
#define HIDC 2048
#define VOCC 32000
#define LSEQ 64
#define NHEAD 8
#define HDIM 64
#define LRATE 3e-4f

__device__ __forceinline__ v8f wmma_bf16(v16bf a, v16bf b, v8f c) {
  return __builtin_amdgcn_wmma_f32_16x16x32_bf16(false, a, false, b, (short)0, c,
                                                 false, false);
}

__device__ __forceinline__ v4bf cvt4(f32x4 q) {
  v4bf w;
  w[0] = (__bf16)q[0]; w[1] = (__bf16)q[1];
  w[2] = (__bf16)q[2]; w[3] = (__bf16)q[3];
  return w;
}

__device__ __forceinline__ v16bf frag16(const __bf16* lo, const __bf16* hi) {
  v8bf l = *(const v8bf*)lo;
  v8bf h = *(const v8bf*)hi;
  return __builtin_shufflevector(l, h, 0, 1, 2, 3, 4, 5, 6, 7,
                                 8, 9, 10, 11, 12, 13, 14, 15);
}

// ---------------------------------------------------------------------------
// LDS-staged, software-pipelined bf16 WMMA GEMM:
//   C = alpha*(op(A)@op(B)) [+ bias] [+ beta*resid]
// Block tile 128(M) x 64(N), K stepped by 32. 256 threads = 8 waves,
// wave (wm,wn) in 4x2 arrangement computes a 32x32 subtile via 2x2 WMMA.
// Requires M%128==0, N%64==0, K%32==0 (true for every GEMM in this model).
// ---------------------------------------------------------------------------
template <bool TA, bool TB, bool HB, bool HR>
__global__ __launch_bounds__(256) void k_gemm(
    const float* __restrict__ A, int lda,
    const float* __restrict__ B, int ldb,
    float* __restrict__ C, int ldc,
    const float* __restrict__ bias,
    const float* __restrict__ resid,
    float alpha, float beta, int K)
{
  __shared__ __bf16 As[128][32];   // [m][k]
  __shared__ __bf16 Bs[64][32];    // transposed: [n][k]

  const int tid  = threadIdx.x;
  const int lane = tid & 31;
  const int wid  = tid >> 5;
  const int tm   = blockIdx.y * 128;
  const int tn   = blockIdx.x * 64;
  const int hl   = lane >> 4;      // half-wave select
  const int lm   = lane & 15;
  const int wm   = (wid & 3) * 32; // wave row offset within block tile
  const int wn   = (wid >> 2) * 32;

  // staging geometry: A tile 128x32 = 1024 float4 (256 thr x 4),
  //                   B tile 32x64  =  512 float4 (256 thr x 2)
  int amr[4], akq[4];
#pragma unroll
  for (int r = 0; r < 4; ++r) { int id = tid + r * 256; amr[r] = id >> 3; akq[r] = (id & 7) * 4; }
  int bi0[2], bi1[2];              // TB: (n, kq) ; !TB: (k, nq)
#pragma unroll
  for (int r = 0; r < 2; ++r) {
    int id = tid + r * 256;
    if (TB) { bi0[r] = id >> 3; bi1[r] = (id & 7) * 4; }
    else    { bi0[r] = id >> 4; bi1[r] = (id & 15) * 4; }
  }

  f32x4 areg[4], breg[2];

  auto load_tile = [&](int k0) {
#pragma unroll
    for (int r = 0; r < 4; ++r) {
      if (TA) {
        const float* p = A + (long)(k0 + akq[r]) * lda + (tm + amr[r]);
        areg[r][0] = p[0]; areg[r][1] = p[(long)lda];
        areg[r][2] = p[2L * lda]; areg[r][3] = p[3L * lda];
      } else {
        areg[r] = *(const f32x4*)(A + (long)(tm + amr[r]) * lda + (k0 + akq[r]));
      }
    }
#pragma unroll
    for (int r = 0; r < 2; ++r) {
      if (TB) breg[r] = *(const f32x4*)(B + (long)(tn + bi0[r]) * ldb + (k0 + bi1[r]));
      else    breg[r] = *(const f32x4*)(B + (long)(k0 + bi0[r]) * ldb + (tn + bi1[r]));
    }
  };

  auto store_tile = [&]() {
#pragma unroll
    for (int r = 0; r < 4; ++r) *(v4bf*)&As[amr[r]][akq[r]] = cvt4(areg[r]);
#pragma unroll
    for (int r = 0; r < 2; ++r) {
      if (TB) {
        *(v4bf*)&Bs[bi0[r]][bi1[r]] = cvt4(breg[r]);
      } else {
        Bs[bi1[r] + 0][bi0[r]] = (__bf16)breg[r][0];
        Bs[bi1[r] + 1][bi0[r]] = (__bf16)breg[r][1];
        Bs[bi1[r] + 2][bi0[r]] = (__bf16)breg[r][2];
        Bs[bi1[r] + 3][bi0[r]] = (__bf16)breg[r][3];
      }
    }
  };

  v8f acc00 = {}, acc01 = {}, acc10 = {}, acc11 = {};
  const int ar0 = wm + lm, ar1 = wm + 16 + lm;
  const int bc0 = wn + lm, bc1 = wn + 16 + lm;

  auto compute = [&]() {
    // A frag (16x32): lane row = m, e=0..7 -> k=hl*8+e ; e=8..15 -> k=16+hl*8+(e-8)
    // B frag (32x16): lane col = n, e=0..7 -> k=hl*8+e ; e=8..15 -> k=16+hl*8+(e-8)
    v16bf a0 = frag16(&As[ar0][hl * 8], &As[ar0][16 + hl * 8]);
    v16bf a1 = frag16(&As[ar1][hl * 8], &As[ar1][16 + hl * 8]);
    v16bf b0 = frag16(&Bs[bc0][hl * 8], &Bs[bc0][16 + hl * 8]);
    v16bf b1 = frag16(&Bs[bc1][hl * 8], &Bs[bc1][16 + hl * 8]);
    acc00 = wmma_bf16(a0, b0, acc00);
    acc01 = wmma_bf16(a0, b1, acc01);
    acc10 = wmma_bf16(a1, b0, acc10);
    acc11 = wmma_bf16(a1, b1, acc11);
  };

  // software pipeline: loads for tile k+1 fly while tile k is in the WMMAs
  load_tile(0);
  store_tile();
  __syncthreads();
  for (int k0 = 32; k0 < K; k0 += 32) {
    load_tile(k0);       // global_load_b128, no wait yet
    compute();           // ds_load + wmma on current LDS tile
    __syncthreads();
    store_tile();        // s_wait_loadcnt happens here
    __syncthreads();
  }
  compute();

  // ---- epilogue: C/D layout lane = column (lm), VGPR v = row (v + hl*8) ----
#pragma unroll
  for (int mi = 0; mi < 2; ++mi) {
#pragma unroll
    for (int ni = 0; ni < 2; ++ni) {
      v8f acc = (mi == 0) ? (ni == 0 ? acc00 : acc01)
                          : (ni == 0 ? acc10 : acc11);
      int col = tn + wn + ni * 16 + lm;
      float bv = HB ? bias[col] : 0.f;
#pragma unroll
      for (int v = 0; v < 8; ++v) {
        int m = tm + wm + mi * 16 + v + hl * 8;
        float r = alpha * acc[v] + bv;
        if (HR) r += beta * resid[(long)m * ldc + col];
        C[(long)m * ldc + col] = r;
      }
    }
  }
}

// Only 4 (TA,TB,bias,resid) combos occur in this model; dispatch exactly those.
static void gemm(hipStream_t s,
                 const float* A, int lda, bool ta,
                 const float* B, int ldb, bool tb,
                 float* C, int ldc,
                 const float* bias, const float* resid,
                 float alpha, float beta,
                 int M, int N, int K)
{
  dim3 grid(N / 64, M / 128), blk(256);
  if (!ta && !tb && bias && !resid)        // fwd: qkv, mlp1, head
    k_gemm<false, false, true, false><<<grid, blk, 0, s>>>(A, lda, B, ldb, C, ldc, bias, resid, alpha, beta, K);
  else if (!ta && !tb)                     // fwd residual: proj, mlp2
    k_gemm<false, false, true, true ><<<grid, blk, 0, s>>>(A, lda, B, ldb, C, ldc, bias, resid, alpha, beta, K);
  else if (!ta && tb)                      // bwd data grads: X @ W^T
    k_gemm<false, true, false, false><<<grid, blk, 0, s>>>(A, lda, B, ldb, C, ldc, bias, resid, alpha, beta, K);
  else                                     // weight update: W -= lr * X^T @ dY
    k_gemm<true, false, false, true ><<<grid, blk, 0, s>>>(A, lda, B, ldb, C, ldc, bias, resid, alpha, beta, K);
}

// ---------------------------------------------------------------------------
// Elementwise / utility kernels
// ---------------------------------------------------------------------------
__global__ void k_copy(const float* __restrict__ src, float* __restrict__ dst, long n) {
  long i = (long)blockIdx.x * blockDim.x + threadIdx.x;
  if (i < n) dst[i] = src[i];
}

__global__ void k_embed(const int* __restrict__ x, const float* __restrict__ emb,
                        const float* __restrict__ pos, float* __restrict__ h, int c) {
  int idx = blockIdx.x * blockDim.x + threadIdx.x;
  if (idx >= NTOK * DIMC) return;
  int n = idx >> 9, d = idx & 511;
  int b = n >> 6, l = n & 63;
  int tok = x[b * 257 + c * LSEQ + l];
  h[idx] = emb[(long)tok * DIMC + d] + pos[l * DIMC + d];
}

__global__ void k_ln_fwd(const float* __restrict__ x, const float* __restrict__ g,
                         const float* __restrict__ b, float* __restrict__ y,
                         float* __restrict__ mean, float* __restrict__ rstd) {
  int row = blockIdx.x, tid = threadIdx.x;
  const float* xr = x + (long)row * DIMC;
  __shared__ float red[256];
  float s = 0.f;
  for (int i = tid; i < DIMC; i += 256) s += xr[i];
  red[tid] = s; __syncthreads();
  for (int o = 128; o > 0; o >>= 1) { if (tid < o) red[tid] += red[tid + o]; __syncthreads(); }
  float mu = red[0] / DIMC; __syncthreads();
  float v = 0.f;
  for (int i = tid; i < DIMC; i += 256) { float d = xr[i] - mu; v += d * d; }
  red[tid] = v; __syncthreads();
  for (int o = 128; o > 0; o >>= 1) { if (tid < o) red[tid] += red[tid + o]; __syncthreads(); }
  float rs = rsqrtf(red[0] / DIMC + 1e-5f);
  if (tid == 0) { mean[row] = mu; rstd[row] = rs; }
  float* yr = y + (long)row * DIMC;
  for (int i = tid; i < DIMC; i += 256) yr[i] = (xr[i] - mu) * rs * g[i] + b[i];
}

__global__ void k_ln_bwd(const float* __restrict__ dy, const float* __restrict__ x,
                         const float* __restrict__ mean, const float* __restrict__ rstd,
                         const float* __restrict__ g, const float* __restrict__ base,
                         float* __restrict__ dx) {
  int row = blockIdx.x, tid = threadIdx.x;
  const float* dyr = dy + (long)row * DIMC;
  const float* xr  = x  + (long)row * DIMC;
  float mu = mean[row], rs = rstd[row];
  __shared__ float r1[256], r2[256];
  float s1 = 0.f, s2 = 0.f;
  for (int i = tid; i < DIMC; i += 256) {
    float dg = dyr[i] * g[i];
    float xh = (xr[i] - mu) * rs;
    s1 += dg; s2 += dg * xh;
  }
  r1[tid] = s1; r2[tid] = s2; __syncthreads();
  for (int o = 128; o > 0; o >>= 1) {
    if (tid < o) { r1[tid] += r1[tid + o]; r2[tid] += r2[tid + o]; }
    __syncthreads();
  }
  s1 = r1[0] / DIMC; s2 = r2[0] / DIMC;
  float* dxr = dx + (long)row * DIMC;
  for (int i = tid; i < DIMC; i += 256) {
    float dg = dyr[i] * g[i];
    float xh = (xr[i] - mu) * rs;
    float v = rs * (dg - s1 - xh * s2);
    if (base) v += base[(long)row * DIMC + i];
    dxr[i] = v;
  }
}

__global__ void k_gelu_fwd(const float* __restrict__ u, float* __restrict__ g, long n) {
  long i = (long)blockIdx.x * blockDim.x + threadIdx.x;
  if (i >= n) return;
  float x = u[i];
  float t = tanhf(0.7978845608028654f * (x + 0.044715f * x * x * x));
  g[i] = 0.5f * x * (1.f + t);
}

__global__ void k_gelu_bwd(const float* __restrict__ u, float* __restrict__ dg, long n) {
  long i = (long)blockIdx.x * blockDim.x + threadIdx.x;
  if (i >= n) return;
  float x = u[i], x2 = x * x;
  float t = tanhf(0.7978845608028654f * (x + 0.044715f * x * x2));
  float der = 0.5f * (1.f + t)
            + 0.5f * x * (1.f - t * t) * 0.7978845608028654f * (1.f + 3.f * 0.044715f * x2);
  dg[i] *= der;
}

__global__ void k_colsum_update(const float* __restrict__ dY, float* __restrict__ bv,
                                int rows, int cols, float lr) {
  int c = blockIdx.x * blockDim.x + threadIdx.x;
  if (c >= cols) return;
  float s = 0.f;
  for (int r = 0; r < rows; ++r) s += dY[(long)r * cols + c];
  bv[c] -= lr * s;
}

// ---------------------------------------------------------------------------
// Attention: per (b,h) block, L=64 rows, HD=64, causal softmax.
// LDS-staged 64x64 tiles (<=64KB/WG). Probabilities saved for backward.
// ---------------------------------------------------------------------------
__global__ void k_attn_fwd(const float* __restrict__ qkv,
                           float* __restrict__ probs, float* __restrict__ o) {
  int bh = blockIdx.x, b = bh / NHEAD, h = bh % NHEAD;
  __shared__ float Qs[64][64], Ks[64][64], Vs[64][64], Ss[64][64];
  int t = threadIdx.x;
  const float* base = qkv + (long)(b * LSEQ + t) * (3 * DIMC) + h * HDIM;
  for (int d = 0; d < HDIM; ++d) {
    Qs[t][d] = base[d];
    Ks[t][d] = base[DIMC + d];
    Vs[t][d] = base[2 * DIMC + d];
  }
  __syncthreads();
  float mx = -1e30f;
  for (int j = 0; j <= t; ++j) {
    float acc = 0.f;
    for (int d = 0; d < HDIM; ++d) acc += Qs[t][d] * Ks[j][d];
    acc *= 0.125f;               // 1/sqrt(64)
    Ss[t][j] = acc;
    mx = fmaxf(mx, acc);
  }
  float sum = 0.f;
  for (int j = 0; j <= t; ++j) { float e = __expf(Ss[t][j] - mx); Ss[t][j] = e; sum += e; }
  float inv = 1.f / sum;
  float* pr = probs + (long)bh * LSEQ * LSEQ + (long)t * LSEQ;
  for (int j = 0; j < LSEQ; ++j) {
    float p = (j <= t) ? Ss[t][j] * inv : 0.f;
    Ss[t][j] = p;
    pr[j] = p;
  }
  float* orow = o + (long)(b * LSEQ + t) * DIMC + h * HDIM;
  for (int d = 0; d < HDIM; ++d) {
    float acc = 0.f;
    for (int j = 0; j <= t; ++j) acc += Ss[t][j] * Vs[j][d];
    orow[d] = acc;
  }
}

__global__ void k_attn_bwd(const float* __restrict__ qkv, const float* __restrict__ probs,
                           const float* __restrict__ dO, float* __restrict__ dqkv) {
  int bh = blockIdx.x, b = bh / NHEAD, h = bh % NHEAD;
  __shared__ float Ks[64][64], Vs[64][64], As[64][64], DSs[64][64];
  int i = threadIdx.x;
  const float* base = qkv + (long)(b * LSEQ + i) * (3 * DIMC) + h * HDIM;
  for (int d = 0; d < HDIM; ++d) { Ks[i][d] = base[DIMC + d]; Vs[i][d] = base[2 * DIMC + d]; }
  const float* prow = probs + (long)bh * LSEQ * LSEQ + (long)i * LSEQ;
  for (int j = 0; j < LSEQ; ++j) As[i][j] = prow[j];
  __syncthreads();
  const float* dorow = dO + (long)(b * LSEQ + i) * DIMC + h * HDIM;
  float ssum = 0.f;
  for (int j = 0; j < LSEQ; ++j) {
    float da = 0.f;
    for (int d = 0; d < HDIM; ++d) da += dorow[d] * Vs[j][d];
    DSs[i][j] = da;
    ssum += As[i][j] * da;
  }
  for (int j = 0; j < LSEQ; ++j) DSs[i][j] = As[i][j] * (DSs[i][j] - ssum);  // masked: A==0
  __syncthreads();
  float* dqrow = dqkv + (long)(b * LSEQ + i) * (3 * DIMC) + h * HDIM;
  for (int d = 0; d < HDIM; ++d) {            // dQ row i
    float acc = 0.f;
    for (int j = 0; j < LSEQ; ++j) acc += DSs[i][j] * Ks[j][d];
    dqrow[d] = 0.125f * acc;
  }
  int j = i;                                   // dK / dV column j
  for (int d = 0; d < HDIM; ++d) {
    float acck = 0.f, accv = 0.f;
    for (int r = 0; r < LSEQ; ++r) {
      float qv  = qkv[(long)(b * LSEQ + r) * (3 * DIMC) + h * HDIM + d];
      float dov = dO[(long)(b * LSEQ + r) * DIMC + h * HDIM + d];
      acck += DSs[r][j] * qv;
      accv += As[r][j] * dov;
    }
    dqrow[DIMC + d]     = 0.125f * acck;
    dqrow[2 * DIMC + d] = accv;
  }
}

// ---------------------------------------------------------------------------
// Loss head helpers
// ---------------------------------------------------------------------------
__global__ void k_emit_logits(const float* __restrict__ logits, float* __restrict__ out, int c) {
  long idx = (long)blockIdx.x * blockDim.x + threadIdx.x;
  if (idx >= (long)NTOK * VOCC) return;
  int n = (int)(idx / VOCC), v = (int)(idx % VOCC);
  int b = n >> 6, l = n & 63;
  out[((long)(b * 256 + c * LSEQ + l)) * VOCC + v] = logits[idx];
}

__global__ void k_xent_grad(float* __restrict__ logits, const int* __restrict__ x, int c) {
  int row = blockIdx.x, tid = threadIdx.x;
  int b = row >> 6, l = row & 63;
  int tgt = x[b * 257 + c * LSEQ + l + 1];
  float* p = logits + (long)row * VOCC;
  __shared__ float red[256];
  float mx = -1e30f;
  for (int i = tid; i < VOCC; i += 256) mx = fmaxf(mx, p[i]);
  red[tid] = mx; __syncthreads();
  for (int o = 128; o > 0; o >>= 1) { if (tid < o) red[tid] = fmaxf(red[tid], red[tid + o]); __syncthreads(); }
  mx = red[0]; __syncthreads();
  float s = 0.f;
  for (int i = tid; i < VOCC; i += 256) s += __expf(p[i] - mx);
  red[tid] = s; __syncthreads();
  for (int o = 128; o > 0; o >>= 1) { if (tid < o) red[tid] += red[tid + o]; __syncthreads(); }
  float inv = 1.f / red[0];
  const float sc = 1.f / 256.f;   // mean over B*MBS targets
  for (int i = tid; i < VOCC; i += 256) {
    float pr = __expf(p[i] - mx) * inv;
    p[i] = (pr - (i == tgt ? 1.f : 0.f)) * sc;
  }
}

// ---------------------------------------------------------------------------
// Host orchestration
// ---------------------------------------------------------------------------
extern "C" void kernel_launch(void* const* d_in, const int* in_sizes, int n_in,
                              void* d_out, int out_size, void* d_ws, size_t ws_size,
                              hipStream_t stream)
{
  (void)in_sizes; (void)n_in; (void)out_size; (void)ws_size;

  const int*   xtok   = (const int*)  d_in[0];
  const float* embedW = (const float*)d_in[1];
  const float* posW   = (const float*)d_in[2];
  const float* ln1_g  = (const float*)d_in[3];
  const float* ln1_b  = (const float*)d_in[4];
  const float* qkv_w  = (const float*)d_in[5];
  const float* qkv_b  = (const float*)d_in[6];
  const float* proj_w = (const float*)d_in[7];
  const float* proj_b = (const float*)d_in[8];
  const float* ln2_g  = (const float*)d_in[9];
  const float* ln2_b  = (const float*)d_in[10];
  const float* mlp_w1 = (const float*)d_in[11];
  const float* mlp_b1 = (const float*)d_in[12];
  const float* mlp_w2 = (const float*)d_in[13];
  const float* mlp_b2 = (const float*)d_in[14];
  const float* lnf_g  = (const float*)d_in[15];
  const float* lnf_b  = (const float*)d_in[16];
  const float* head_w = (const float*)d_in[17];
  const float* head_b = (const float*)d_in[18];
  float* out = (float*)d_out;

  // ---- workspace layout (floats) ----
  float* W = (float*)d_ws;
  size_t off = 0;
  auto alloc = [&](long n) { float* p = W + off; off += (size_t)n; return p; };

  const long TD = (long)NTOK * DIMC;       // 131072
  const long TH = (long)NTOK * HIDC;       // 524288
  const long TQ = (long)NTOK * 3 * DIMC;   // 393216
  const long PP = (long)4 * NHEAD * LSEQ * LSEQ; // 131072

  float* w1c = alloc(4L * DIMC * HIDC);
  float* b1c = alloc(4L * HIDC);
  float* w2c = alloc(4L * HIDC * DIMC);
  float* b2c = alloc(4L * DIMC);
  float* hb[5];   for (int i = 0; i < 5; ++i) hb[i]   = alloc(TD);
  float* hmid[4]; for (int i = 0; i < 4; ++i) hmid[i] = alloc(TD);
  float* xn2s[4]; for (int i = 0; i < 4; ++i) xn2s[i] = alloc(TD);
  float* us[4];   for (int i = 0; i < 4; ++i) us[i]   = alloc(TH);
  float* gs[4];   for (int i = 0; i < 4; ++i) gs[i]   = alloc(TH);
  float* qkvs[4]; for (int i = 0; i < 4; ++i) qkvs[i] = alloc(TQ);
  float* prb[4];  for (int i = 0; i < 4; ++i) prb[i]  = alloc(PP);
  float* mean1 = alloc(4 * NTOK); float* rstd1 = alloc(4 * NTOK);
  float* mean2 = alloc(4 * NTOK); float* rstd2 = alloc(4 * NTOK);
  float* meanf = alloc(NTOK);     float* rstdf = alloc(NTOK);
  float* xn_t  = alloc(TD);
  float* o_t   = alloc(TD);
  float* hn    = alloc(TD);
  float* logits = alloc((long)NTOK * VOCC);
  float* dG    = alloc(TH);
  float* dxn2  = alloc(TD);
  float* d_o   = alloc(TD);
  float* dqkv  = alloc(TQ);
  float* dxn   = alloc(TD);
  float* dbufA = alloc(TD);
  float* dbufB = alloc(TD);

  // ---- copy trainable (dyn) weights into ws; updated across chunks ----
  {
    long nw = 4L * DIMC * HIDC;
    k_copy<<<(int)((nw + 255) / 256), 256, 0, stream>>>(mlp_w1, w1c, nw);
    k_copy<<<(int)((4L * HIDC + 255) / 256), 256, 0, stream>>>(mlp_b1, b1c, 4L * HIDC);
    k_copy<<<(int)((nw + 255) / 256), 256, 0, stream>>>(mlp_w2, w2c, nw);
    k_copy<<<(int)((4L * DIMC + 255) / 256), 256, 0, stream>>>(mlp_b2, b2c, 4L * DIMC);
  }

  for (int c = 0; c < 4; ++c) {
    // ======================= forward =======================
    k_embed<<<512, 256, 0, stream>>>(xtok, embedW, posW, hb[0], c);
    for (int i = 0; i < 4; ++i) {
      k_ln_fwd<<<NTOK, 256, 0, stream>>>(hb[i], ln1_g + i * DIMC, ln1_b + i * DIMC,
                                         xn_t, mean1 + i * NTOK, rstd1 + i * NTOK);
      gemm(stream, xn_t, DIMC, false, qkv_w + (long)i * DIMC * 3 * DIMC, 3 * DIMC, false,
           qkvs[i], 3 * DIMC, qkv_b + i * 3 * DIMC, nullptr, 1.f, 0.f,
           NTOK, 3 * DIMC, DIMC);
      k_attn_fwd<<<32, 64, 0, stream>>>(qkvs[i], prb[i], o_t);
      gemm(stream, o_t, DIMC, false, proj_w + (long)i * DIMC * DIMC, DIMC, false,
           hmid[i], DIMC, proj_b + i * DIMC, hb[i], 1.f, 1.f, NTOK, DIMC, DIMC);
      k_ln_fwd<<<NTOK, 256, 0, stream>>>(hmid[i], ln2_g + i * DIMC, ln2_b + i * DIMC,
                                         xn2s[i], mean2 + i * NTOK, rstd2 + i * NTOK);
      gemm(stream, xn2s[i], DIMC, false, w1c + (long)i * DIMC * HIDC, HIDC, false,
           us[i], HIDC, b1c + i * HIDC, nullptr, 1.f, 0.f, NTOK, HIDC, DIMC);
      k_gelu_fwd<<<(int)((TH + 255) / 256), 256, 0, stream>>>(us[i], gs[i], TH);
      gemm(stream, gs[i], HIDC, false, w2c + (long)i * HIDC * DIMC, DIMC, false,
           hb[i + 1], DIMC, b2c + i * DIMC, hmid[i], 1.f, 1.f, NTOK, DIMC, HIDC);
    }
    k_ln_fwd<<<NTOK, 256, 0, stream>>>(hb[4], lnf_g, lnf_b, hn, meanf, rstdf);
    gemm(stream, hn, DIMC, false, head_w, VOCC, false,
         logits, VOCC, head_b, nullptr, 1.f, 0.f, NTOK, VOCC, DIMC);
    k_emit_logits<<<VOCC, 256, 0, stream>>>(logits, out, c);
    k_xent_grad<<<NTOK, 256, 0, stream>>>(logits, xtok, c);   // logits -> dlogits in place

    // ======================= backward =======================
    float* dh = dbufA; float* dtmp = dbufB;
    gemm(stream, logits, VOCC, false, head_w, VOCC, true,
         dh, DIMC, nullptr, nullptr, 1.f, 0.f, NTOK, DIMC, VOCC);       // dhn
    k_ln_bwd<<<NTOK, 256, 0, stream>>>(dh, hb[4], meanf, rstdf, lnf_g, nullptr, dtmp);
    { float* t = dh; dh = dtmp; dtmp = t; }

    for (int i = 3; i >= 0; --i) {
      // MLP backward (+ fused SGD weight updates; dG/dxn2 use pre-update weights)
      gemm(stream, dh, DIMC, false, w2c + (long)i * HIDC * DIMC, DIMC, true,
           dG, HIDC, nullptr, nullptr, 1.f, 0.f, NTOK, HIDC, DIMC);      // dg = dh @ w2^T
      k_colsum_update<<<2, 256, 0, stream>>>(dh, b2c + i * DIMC, NTOK, DIMC, LRATE);
      gemm(stream, gs[i], HIDC, true, dh, DIMC, false,
           w2c + (long)i * HIDC * DIMC, DIMC, nullptr, w2c + (long)i * HIDC * DIMC,
           -LRATE, 1.f, HIDC, DIMC, NTOK);                               // w2 -= lr*g^T@dh
      k_gelu_bwd<<<(int)((TH + 255) / 256), 256, 0, stream>>>(us[i], dG, TH); // du
      k_colsum_update<<<8, 256, 0, stream>>>(dG, b1c + i * HIDC, NTOK, HIDC, LRATE);
      gemm(stream, dG, HIDC, false, w1c + (long)i * DIMC * HIDC, HIDC, true,
           dxn2, DIMC, nullptr, nullptr, 1.f, 0.f, NTOK, DIMC, HIDC);    // dxn2 = du @ w1^T
      gemm(stream, xn2s[i], DIMC, true, dG, HIDC, false,
           w1c + (long)i * DIMC * HIDC, HIDC, nullptr, w1c + (long)i * DIMC * HIDC,
           -LRATE, 1.f, DIMC, HIDC, NTOK);                               // w1 -= lr*xn2^T@du
      k_ln_bwd<<<NTOK, 256, 0, stream>>>(dxn2, hmid[i], mean2 + i * NTOK, rstd2 + i * NTOK,
                                         ln2_g + i * DIMC, dh, dtmp);    // dh_mid
      { float* t = dh; dh = dtmp; dtmp = t; }

      // Attention backward (static weights: only propagate activations)
      gemm(stream, dh, DIMC, false, proj_w + (long)i * DIMC * DIMC, DIMC, true,
           d_o, DIMC, nullptr, nullptr, 1.f, 0.f, NTOK, DIMC, DIMC);     // do
      k_attn_bwd<<<32, 64, 0, stream>>>(qkvs[i], prb[i], d_o, dqkv);
      gemm(stream, dqkv, 3 * DIMC, false, qkv_w + (long)i * DIMC * 3 * DIMC, 3 * DIMC, true,
           dxn, DIMC, nullptr, nullptr, 1.f, 0.f, NTOK, DIMC, 3 * DIMC); // dxn
      k_ln_bwd<<<NTOK, 256, 0, stream>>>(dxn, hb[i], mean1 + i * NTOK, rstd1 + i * NTOK,
                                         ln1_g + i * DIMC, dh, dtmp);    // dh_in
      { float* t = dh; dh = dtmp; dtmp = t; }
    }
  }
}
